// EngramModule_2937757630928
// MI455X (gfx1250) — matrix operand order repaired and checked by reading, not compiled, not run on hardware
//
#include <hip/hip_runtime.h>
#include <stdint.h>

#define T_SEQ 4096
#define B_SZ  4
#define H_DIM 2048
#define NH    16
#define DH    64
#define E_DIM 1024

typedef __attribute__((ext_vector_type(16))) __bf16 v16bf;
typedef __attribute__((ext_vector_type(8)))  float  v8f;
typedef __attribute__((ext_vector_type(4)))  unsigned int v4u;
typedef __attribute__((ext_vector_type(8)))  int    v8i;
typedef __attribute__((ext_vector_type(4)))  int    v4i;

#if __has_builtin(__builtin_amdgcn_tensor_load_to_lds) && \
    __has_builtin(__builtin_amdgcn_s_wait_tensorcnt)
#define HAVE_TDM 1
#else
#define HAVE_TDM 0
#endif

__device__ __forceinline__ uint32_t f2bf_pack(float lo, float hi) {
  uint32_t a = __float_as_uint(lo);
  uint32_t b = __float_as_uint(hi);
  a = a + 0x7fffu + ((a >> 16) & 1u);   // RNE to bf16
  b = b + 0x7fffu + ((b >> 16) & 1u);
  return (a >> 16) | (b & 0xffff0000u);
}

#if HAVE_TDM
// TDM: copy 1KB (256 dwords, 1-D tile) from global to LDS. D# per ISA ch.8:
// group0: count=1 | lds_addr | global_addr[56:0] | type=2
// group1: data_size=4B, tensor_dim0=256, tile_dim0=256, dim0_stride=256
// 6-arg toolchain variant: (v4u g0, v8i g1, v4i g2, v4i g3, v8i g4, i32 cpol)
__device__ __forceinline__ void tdm_load_1kb(uint32_t lds_off,
                                             const uint32_t* gsrc) {
  uint64_t ga = (uint64_t)(uintptr_t)gsrc;
  v4u g0;
  g0.x = 1u;                                        // count=1 (valid), user mode
  g0.y = lds_off;                                   // lds_addr (bytes)
  g0.z = (uint32_t)(ga & 0xffffffffu);              // global_addr[31:0]
  g0.w = (uint32_t)((ga >> 32) & 0x1ffffffu) | (2u << 30);  // [56:32] | type=2
  v8i g1;
  g1[0] = 0x20000;                                  // data_size=2 (4B), mask=0
  g1[1] = (int)(256u << 16);                        // tensor_dim0[15:0] @ [63:48]
  g1[2] = 0;                                        // tensor_dim0 hi, tensor_dim1 lo
  g1[3] = (int)(256u << 16);                        // tile_dim0=256 @ [127:112]
  g1[4] = 0;                                        // tile_dim1=0 (1-D), tile_dim2=0
  g1[5] = 256;                                      // tensor_dim0_stride
  g1[6] = 0;
  g1[7] = 0;
  v4i z4 = {0, 0, 0, 0};
  v8i z8 = {0, 0, 0, 0, 0, 0, 0, 0};
  __builtin_amdgcn_tensor_load_to_lds(g0, g1, z4, z4, z8, 0);
}
#endif

// ---------------------------------------------------------------------------
// Kernel 1: pack key_W (4 groups) + value_W into bf16 WMMA B-fragment layout.
// B frag (16-bit, 32x16): lanes 0-15 hold K={0..7}u{16..23} (col n=lane&15),
// lanes 16-31 hold K={8..15}u{24..31}; 2 k's per dword.
// Wpack[((m*32+kt)*128+nt)*256 + lane*8 + d]
// ---------------------------------------------------------------------------
__global__ __launch_bounds__(256) void pack_w(const float* __restrict__ keyW,
                                              const float* __restrict__ valW,
                                              uint32_t* __restrict__ Wpack) {
  size_t idx = (size_t)blockIdx.x * 256 + threadIdx.x;   // < 5 * 2^20
  uint32_t d    = idx & 7u;
  uint32_t lane = (idx >> 3) & 31u;
  uint32_t nt   = (idx >> 8) & 127u;
  uint32_t kt   = (idx >> 15) & 31u;
  uint32_t m    = (uint32_t)(idx >> 20);                 // 0..3 key groups, 4 value
  int n = nt * 16 + (lane & 15);
  int k = kt * 32 + ((lane >> 4) << 3) + ((d < 4) ? 2 * d : 16 + 2 * (d - 4));
  const float* src = (m < 4) ? (keyW + ((size_t)m * H_DIM + n) * E_DIM + k)
                             : (valW + (size_t)n * E_DIM + k);
  Wpack[idx] = f2bf_pack(src[0], src[1]);
}

// ---------------------------------------------------------------------------
// Kernel 2: hashed gather -> bf16 WMMA A-fragment layout.
// A frag (16-bit, 16x32): lane<16: row=lane, k in {0..7}u{16..23};
// lane>=16: row=lane-16, k in {8..15}u{24..31}; 2 k's per dword.
// Apack[((mt*32+kt)*32+lane)*8 + d]
// ---------------------------------------------------------------------------
__global__ __launch_bounds__(256) void gather_a(const int* __restrict__ hash_ids,
                                                const float* __restrict__ emb,
                                                uint32_t* __restrict__ Apack) {
  size_t idx = (size_t)blockIdx.x * 256 + threadIdx.x;   // < 2^23
  uint32_t d    = idx & 7u;
  uint32_t lane = (idx >> 3) & 31u;
  uint32_t kt   = (idx >> 8) & 31u;
  uint32_t mt   = (uint32_t)(idx >> 13);
  int row = mt * 16 + (lane & 15);
  int k = kt * 32 + ((lane >> 4) << 3) + ((d < 4) ? 2 * d : 16 + 2 * (d - 4));
  int head = k >> 6;
  int id = hash_ids[(size_t)row * NH + head];
  size_t base = ((size_t)(id + head * 100000)) * DH + (k & 63);
  Apack[idx] = f2bf_pack(emb[base], emb[base + 1]);
}

// ---------------------------------------------------------------------------
// Kernel 3: fused GEMM. grid = (1024 M-tiles, 5 matrices). 512 threads = 16
// waves; wave w owns cols [w*128, w*128+128). K = 1024 (32 x k32 steps).
// A fragment (1KB, identical for all 16 waves) is TDM-staged into
// double-buffered LDS; B fragments are double-buffered in registers so the
// WMMAs of step kt overlap the loads of step kt+1.
// m<4: keys group m -> gate stats epilogue. m==4: value proj -> vraw + msq.
// ---------------------------------------------------------------------------
__global__ __launch_bounds__(512) void gemm_fused(
    const uint32_t* __restrict__ Apack, const uint32_t* __restrict__ Wpack,
    const float* __restrict__ hs,   const float* __restrict__ keyB,
    const float* __restrict__ n1w,  const float* __restrict__ n2w,
    const float* __restrict__ valB,
    float* __restrict__ gateBuf, float* __restrict__ vraw,
    float* __restrict__ rowmsq) {
  const int mt   = blockIdx.x;
  const int m    = blockIdx.y;
  const int tid  = threadIdx.x;
  const int lane = tid & 31;
  const int wave = tid >> 5;

  __shared__ __align__(32) uint32_t sA[2][256];

  v8f c[8];
#pragma unroll
  for (int i = 0; i < 8; ++i) {
#pragma unroll
    for (int j = 0; j < 8; ++j) c[i][j] = 0.0f;
  }

  const uint32_t* aGlob = Apack + (size_t)mt * 8192;
  const uint32_t* wBase = Wpack + (size_t)m * (32u * 128u * 256u)
                                + (size_t)wave * 2048 + lane * 8;

  // stage A(kt=0) into sA[0]
#if HAVE_TDM
  if (wave == 0) {
    tdm_load_1kb((uint32_t)(uintptr_t)(&sA[0][0]), aGlob);
    __builtin_amdgcn_s_wait_tensorcnt((short)0);
  }
#else
  if (tid < 256) sA[0][tid] = aGlob[tid];
#endif
  __syncthreads();

  // preload B(kt=0)
  v16bf bb[2][8];
#pragma unroll
  for (int i = 0; i < 8; ++i)
    bb[0][i] = *reinterpret_cast<const v16bf*>(wBase + i * 256);

#pragma unroll 2
  for (int kt = 0; kt < 32; ++kt) {
    const int cur = kt & 1;
    if (kt < 31) {
      // issue next A stage (TDM) + next B fragment loads
#if HAVE_TDM
      if (wave == 0)
        tdm_load_1kb((uint32_t)(uintptr_t)(&sA[cur ^ 1][0]),
                     aGlob + (size_t)(kt + 1) * 256);
#else
      if (tid < 256) sA[cur ^ 1][tid] = aGlob[(size_t)(kt + 1) * 256 + tid];
#endif
      const uint32_t* wp = wBase + (size_t)(kt + 1) * 32768;
#pragma unroll
      for (int i = 0; i < 8; ++i)
        bb[cur ^ 1][i] = *reinterpret_cast<const v16bf*>(wp + i * 256);
    }
    v16bf a = *reinterpret_cast<const v16bf*>(&sA[cur][lane * 8]);
#pragma unroll
    for (int i = 0; i < 8; ++i)
      c[i] = __builtin_amdgcn_wmma_f32_16x16x32_bf16(
          false, a, false, bb[cur][i], (short)0, c[i], false, false);
    if (kt < 31) {
#if HAVE_TDM
      if (wave == 0) __builtin_amdgcn_s_wait_tensorcnt((short)0);
#endif
      __syncthreads();
    }
  }

  __shared__ float sred[3][16];
  if (tid < 48) (&sred[0][0])[tid] = 0.0f;
  __syncthreads();

  const int rowBase = (lane < 16) ? 0 : 8;
  const int colLane = lane & 15;
  const int hBase   = wave * 128;

  float s0[8], s1[8], s2[8];
#pragma unroll
  for (int j = 0; j < 8; ++j) { s0[j] = 0.f; s1[j] = 0.f; s2[j] = 0.f; }

  if (m < 4) {
#pragma unroll
    for (int i = 0; i < 8; ++i) {
      int h = hBase + i * 16 + colLane;
      float kb  = keyB[m * H_DIM + h];
      float n12 = n1w[m * H_DIM + h] * n2w[m * H_DIM + h];
#pragma unroll
      for (int j = 0; j < 8; ++j) {
        int bt = mt * 16 + rowBase + j;
        int b = bt >> 12, t = bt & 4095;
        float kv = c[i][j] + kb;
        float hv = hs[((size_t)(t * B_SZ + b)) * H_DIM + h];
        s0[j] += kv * kv;
        s1[j] += hv * hv;
        s2[j] += kv * hv * n12;
      }
    }
  } else {
#pragma unroll
    for (int i = 0; i < 8; ++i) {
      int h = hBase + i * 16 + colLane;
      float vb = valB[h];
#pragma unroll
      for (int j = 0; j < 8; ++j) {
        int bt = mt * 16 + rowBase + j;
        float v = c[i][j] + vb;
        vraw[(size_t)bt * H_DIM + h] = v;
        s0[j] += v * v;
      }
    }
  }

#pragma unroll
  for (int j = 0; j < 8; ++j) {
#pragma unroll
    for (int off = 1; off < 16; off <<= 1) {
      s0[j] += __shfl_xor(s0[j], off, 16);
      s1[j] += __shfl_xor(s1[j], off, 16);
      s2[j] += __shfl_xor(s2[j], off, 16);
    }
  }
  if (colLane == 0) {
#pragma unroll
    for (int j = 0; j < 8; ++j) {
      atomicAdd(&sred[0][rowBase + j], s0[j]);
      if (m < 4) {
        atomicAdd(&sred[1][rowBase + j], s1[j]);
        atomicAdd(&sred[2][rowBase + j], s2[j]);
      }
    }
  }
  __syncthreads();

  if (tid < 16) {
    int bt = mt * 16 + tid;
    if (m < 4) {
      float msk = sred[0][tid] * (1.0f / H_DIM);
      float msq = sred[1][tid] * (1.0f / H_DIM);
      float crs = sred[2][tid];
      float g = crs * rsqrtf(msk + 1.1920929e-07f)
                    * rsqrtf(msq + 1.1920929e-07f) * 0.02209708691f; // 1/sqrt(2048)
      float sg = (g > 0.f) ? 1.f : ((g < 0.f) ? -1.f : 0.f);
      float gs = sqrtf(fmaxf(fabsf(g), 1e-6f)) * sg;
      gateBuf[(size_t)bt * 4 + m] = 1.0f / (1.0f + expf(-gs));
    } else {
      rowmsq[bt] = sred[0][tid] * (1.0f / H_DIM);
    }
  }
}

// ---------------------------------------------------------------------------
// Kernel 4: per-(b,t) elementwise finish: value = gate*vraw; causal dilated
// depthwise conv over taps t-9,t-6,t-3,t via rms scalars; SiLU; group mean.
// ---------------------------------------------------------------------------
__global__ __launch_bounds__(256) void finalize(
    const float* __restrict__ gateBuf, const float* __restrict__ rowmsq,
    const float* __restrict__ vraw,    const float* __restrict__ scw,
    const float* __restrict__ convw,   float* __restrict__ out) {
  const int bt = blockIdx.y;
  const int b = bt >> 12, t = bt & 4095;
  const int h = blockIdx.x * 256 + threadIdx.x;

  __shared__ float s_gate[4][4];   // [tap][g]
  __shared__ float s_invr[4][4];
  __shared__ int   s_tau[4];

  if (threadIdx.x < 16) {
    int k = threadIdx.x >> 2, g = threadIdx.x & 3;
    int tau = t - 9 + 3 * k;                 // tap 3 == t
    int btk = b * T_SEQ + tau;
    float gv = 0.f, iv = 0.f;
    if (tau >= 0) {
      gv = gateBuf[(size_t)btk * 4 + g];
      float msq = rowmsq[btk];
      iv = rsqrtf(gv * gv * msq + 1e-5f);    // rms(value) via gate^2*mean(vraw^2)
    }
    s_gate[k][g] = gv;
    s_invr[k][g] = iv;
    if (g == 0) s_tau[k] = (tau >= 0) ? btk : -1;
  }
  __syncthreads();

  float vr[4];
#pragma unroll
  for (int k = 0; k < 4; ++k) {
    int btk = s_tau[k];
    vr[k] = (btk >= 0) ? vraw[(size_t)btk * H_DIM + h] : 0.f;
  }

  float acc = 0.f;
#pragma unroll
  for (int g = 0; g < 4; ++g) {
    float sc = scw[g * H_DIM + h];
    const float4 cw =
        *reinterpret_cast<const float4*>(convw + ((size_t)(g * H_DIM + h)) * 4);
    float y = cw.x * (s_gate[0][g] * vr[0] * s_invr[0][g] * sc)
            + cw.y * (s_gate[1][g] * vr[1] * s_invr[1][g] * sc)
            + cw.z * (s_gate[2][g] * vr[2] * s_invr[2][g] * sc)
            + cw.w * (s_gate[3][g] * vr[3] * s_invr[3][g] * sc);
    y = y / (1.0f + expf(-y));               // SiLU
    acc += s_gate[3][g] * vr[3] + y;         // value + conv branch
  }
  out[((size_t)(t * B_SZ + b)) * H_DIM + h] = acc * 0.25f;
}

// ---------------------------------------------------------------------------
extern "C" void kernel_launch(void* const* d_in, const int* in_sizes, int n_in,
                              void* d_out, int out_size, void* d_ws,
                              size_t ws_size, hipStream_t stream) {
  (void)in_sizes; (void)n_in; (void)out_size; (void)ws_size;
  const float* hs   = (const float*)d_in[0];
  const int*   hid  = (const int*)d_in[1];
  const float* emb  = (const float*)d_in[2];
  const float* keyW = (const float*)d_in[3];
  const float* keyB = (const float*)d_in[4];
  const float* valW = (const float*)d_in[5];
  const float* valB = (const float*)d_in[6];
  const float* n1w  = (const float*)d_in[7];
  const float* n2w  = (const float*)d_in[8];
  const float* scw  = (const float*)d_in[9];
  const float* cvw  = (const float*)d_in[10];
  float* out = (float*)d_out;

  char* ws = (char*)d_ws;
  uint32_t* Wpack   = (uint32_t*)(ws);                       // 20,971,520 B
  uint32_t* Apack   = (uint32_t*)(ws + 20971520u);           // 33,554,432 B
  float*    gateBuf = (float*)   (ws + 54525952u);           //    262,144 B
  float*    rowmsq  = (float*)   (ws + 54788096u);           //     65,536 B
  float*    vraw    = (float*)   (ws + 54853632u);           // 134,217,728 B
                                                             // total ~180 MB

  pack_w  <<<20480, 256, 0, stream>>>(keyW, valW, Wpack);
  gather_a<<<32768, 256, 0, stream>>>(hid, emb, Apack);
  gemm_fused<<<dim3(1024, 5), 512, 0, stream>>>(Apack, Wpack, hs, keyB, n1w,
                                                n2w, valB, gateBuf, vraw,
                                                rowmsq);
  finalize<<<dim3(8, 16384), 256, 0, stream>>>(gateBuf, rowmsq, vraw, scw,
                                               cvw, out);
}
